// JKNet_70188355551831
// MI455X (gfx1250) — compile-verified, matrix operand order checked by reading, not run on hardware
//
#include <hip/hip_runtime.h>
#include <hip/hip_bf16.h>
#include <math.h>

typedef __attribute__((ext_vector_type(16))) _Float16 v16h;
typedef __attribute__((ext_vector_type(8)))  _Float16 v8h;
typedef __attribute__((ext_vector_type(8)))  float    v8f;
typedef __attribute__((ext_vector_type(4)))  int      v4i;

#define NFEAT   128
#define NCLASS  40
#define NLAYERS 6
#define AF_STRIDE 136   // 128 + 8 halfword pad -> lane stride 272B, conflict-free b128 reads

#define AS_GLOBAL __attribute__((address_space(1)))
#define AS_LOCAL  __attribute__((address_space(3)))

#if defined(__has_builtin)
#if __has_builtin(__builtin_amdgcn_global_load_async_to_lds_b128) && \
    __has_builtin(__builtin_amdgcn_s_wait_asynccnt)
#define HAVE_ASYNC_LDS 1
#endif
#endif

// ---------------- degree / normalization ----------------

__global__ void k_init_deg(int* __restrict__ deg, int n) {
    int i = blockIdx.x * blockDim.x + threadIdx.x;
    if (i < n) deg[i] = 1;                       // self-loop
}

__global__ void k_deg_accum(const int* __restrict__ dst, int* __restrict__ deg, int E) {
    int e = blockIdx.x * blockDim.x + threadIdx.x;
    if (e < E)
        __hip_atomic_fetch_add(&deg[dst[e]], 1, __ATOMIC_RELAXED, __HIP_MEMORY_SCOPE_AGENT);
}

__global__ void k_dinv(const int* __restrict__ deg, float* __restrict__ dinv, int n) {
    int i = blockIdx.x * blockDim.x + threadIdx.x;
    if (i < n) dinv[i] = rsqrtf((float)deg[i]);  // deg >= 1 always (self-loop)
}

__global__ void k_zero4(float4* __restrict__ p, int n4) {
    int i = blockIdx.x * blockDim.x + threadIdx.x;
    if (i < n4) p[i] = make_float4(0.f, 0.f, 0.f, 0.f);
}

// ---------------- dense transform: T = H @ W + b, Hout = dinv^2 * T ----------------
// Block = 256 threads (8 waves), 64 rows per block. Wave w owns cols [16w,16w+16).
// B fragments (one W col-slice, all K) held in registers, amortized over 4 M-tiles.
// A tile staged in LDS as f16 with padded stride; per k-step: 2x ds_load_b128 + WMMA.

__global__ __launch_bounds__(256)
void k_gemm_wmma(const float* __restrict__ H, const float* __restrict__ W,
                 const float* __restrict__ bias, const float* __restrict__ dinv,
                 float* __restrict__ T, float* __restrict__ Hout, int nrows) {
    __shared__ _Float16 Af[64 * AF_STRIDE];      // ~17 KB
    const int tid = threadIdx.x;
    const int m0  = blockIdx.x * 64;

    // ---- stage A tile (64 x 128) as f16; each thread converts one quarter-row ----
    {
        int r  = tid >> 2;                       // 0..63
        int c0 = (tid & 3) * 32;                 // 0,32,64,96
        int gr = m0 + r; if (gr >= nrows) gr = nrows - 1;
        const float* src = H + (size_t)gr * NFEAT + c0;
        _Float16* dst = Af + r * AF_STRIDE + c0;
#pragma unroll
        for (int j = 0; j < 32; j += 8) {
            float4 f0 = *(const float4*)(src + j);
            float4 f1 = *(const float4*)(src + j + 4);
            v8h h8 = { (_Float16)f0.x, (_Float16)f0.y, (_Float16)f0.z, (_Float16)f0.w,
                       (_Float16)f1.x, (_Float16)f1.y, (_Float16)f1.z, (_Float16)f1.w };
            *(v8h*)(dst + j) = h8;
        }
    }

    const int wave = tid >> 5;
    const int lane = tid & 31;
    const int half = lane >> 4;                  // ISA: lanes 16-31 hold the +8 K-offset half
    const int lrow = lane & 15;
    const int col  = wave * 16 + lrow;           // N index owned by this lane

    // ---- B fragments in registers: gather W columns (L0/L2-resident), f32->f16 ----
    v16h bfrag[4];
    const float* wc = W + col;
#pragma unroll
    for (int ks = 0; ks < 4; ++ks) {
        const int kbase = ks * 32 + half * 8;
#pragma unroll
        for (int e = 0; e < 8; ++e) {
            bfrag[ks][e]     = (_Float16)wc[(kbase + e) * NFEAT];
            bfrag[ks][e + 8] = (_Float16)wc[(kbase + 16 + e) * NFEAT];
        }
    }

    __syncthreads();

    const float bb = bias[col];
#pragma unroll
    for (int mt = 0; mt < 4; ++mt) {
        const _Float16* ar = Af + (mt * 16 + lrow) * AF_STRIDE;
        v8f c = {};
#pragma unroll
        for (int ks = 0; ks < 4; ++ks) {
            const int kbase = ks * 32 + half * 8;
            v8h lo = *(const v8h*)(ar + kbase);          // K = kbase..kbase+7
            v8h hi = *(const v8h*)(ar + kbase + 16);     // K = kbase+16..kbase+23
            v16h a = __builtin_shufflevector(lo, hi, 0,1,2,3,4,5,6,7,
                                                     8,9,10,11,12,13,14,15);
            c = __builtin_amdgcn_wmma_f32_16x16x32_f16(false, a, false, bfrag[ks],
                                                       (short)0, c, false, false);
        }
#pragma unroll
        for (int r = 0; r < 8; ++r) {
            const int m = m0 + mt * 16 + r + half * 8;   // C/D: VGPR r -> M=r (+8 half 1)
            if (m < nrows) {
                float t = c[r] + bb;
                T[(size_t)m * NFEAT + col] = t;
                float dv = dinv[m];
                Hout[(size_t)m * NFEAT + col] = dv * dv * t;   // fused self-loop seed
            }
        }
    }
}

// ---------------- edge scatter: Hout[dst] += dinv[s]*dinv[d] * T[src] ----------------
// One wave per edge; edge scalars are wave-uniform -> scalar loads. Gather + atomics
// are L2-resident (feature matrix = 51 MB < 192 MB L2).

__global__ __launch_bounds__(256)
void k_scatter(const float* __restrict__ T, const int* __restrict__ src,
               const int* __restrict__ dst, const float* __restrict__ dinv,
               float* __restrict__ Hout, int E) {
    const int lane = threadIdx.x & 31;
    const int wv   = __builtin_amdgcn_readfirstlane((int)(threadIdx.x >> 5));
    const int e    = blockIdx.x * 8 + wv;
    if (e >= E) return;
    const int s = src[e];
    const int d = dst[e];
    const float nrm = dinv[s] * dinv[d];
    const float4 v = *(const float4*)(T + (size_t)s * NFEAT + lane * 4);
    float* o = Hout + (size_t)d * NFEAT + lane * 4;
    __hip_atomic_fetch_add(o + 0, nrm * v.x, __ATOMIC_RELAXED, __HIP_MEMORY_SCOPE_AGENT);
    __hip_atomic_fetch_add(o + 1, nrm * v.y, __ATOMIC_RELAXED, __HIP_MEMORY_SCOPE_AGENT);
    __hip_atomic_fetch_add(o + 2, nrm * v.z, __ATOMIC_RELAXED, __HIP_MEMORY_SCOPE_AGENT);
    __hip_atomic_fetch_add(o + 3, nrm * v.w, __ATOMIC_RELAXED, __HIP_MEMORY_SCOPE_AGENT);
}

// ---------------- ReLU in place + running JK max ----------------
__global__ void k_relu_jk(float4* __restrict__ H, float4* __restrict__ JK, int n4) {
    int i = blockIdx.x * blockDim.x + threadIdx.x;
    if (i >= n4) return;
    float4 h = H[i];
    h.x = fmaxf(h.x, 0.f); h.y = fmaxf(h.y, 0.f);
    h.z = fmaxf(h.z, 0.f); h.w = fmaxf(h.w, 0.f);
    H[i] = h;
    float4 j = JK[i];
    j.x = fmaxf(j.x, h.x); j.y = fmaxf(j.y, h.y);
    j.z = fmaxf(j.z, h.z); j.w = fmaxf(j.w, h.w);
    JK[i] = j;
}

// ---------------- FC + log_softmax (block of 64 per node) ----------------
// Row staging uses the CDNA5 async Global->LDS engine when available
// (global_load_async_to_lds_b128, ASYNCcnt), falling back to plain loads.
__global__ __launch_bounds__(64)
void k_fc_lsm(const float* __restrict__ JK, const float* __restrict__ fcW,
              const float* __restrict__ fcb, float* __restrict__ out, int n) {
    __shared__ float row[NFEAT];
    __shared__ float logits[NCLASS];
    __shared__ float red[2];
    const int node = blockIdx.x;
    const int tid  = threadIdx.x;
    if (node >= n) return;
#if defined(HAVE_ASYNC_LDS)
    if (tid < 32) {                               // wave 0: 32 lanes x 16B = 512B row
        const AS_GLOBAL float* gf =
            (const AS_GLOBAL float*)(JK + (size_t)node * NFEAT + tid * 4);
        AS_GLOBAL v4i* g = (AS_GLOBAL v4i*)gf;    // builtin wants non-const v4i pointers
        AS_LOCAL  float* lf = (AS_LOCAL float*)(row + tid * 4);
        AS_LOCAL  v4i* l = (AS_LOCAL v4i*)lf;
        __builtin_amdgcn_global_load_async_to_lds_b128(g, l, 0, 0);
    }
    __builtin_amdgcn_s_wait_asynccnt(0);
    __syncthreads();
#else
    row[tid]      = JK[(size_t)node * NFEAT + tid];
    row[tid + 64] = JK[(size_t)node * NFEAT + 64 + tid];
    __syncthreads();
#endif
    if (tid < NCLASS) {
        float acc = fcb[tid];
#pragma unroll 8
        for (int k = 0; k < NFEAT; ++k) acc += row[k] * fcW[k * NCLASS + tid];
        logits[tid] = acc;
    }
    __syncthreads();
    if (tid == 0) {
        float m = logits[0];
        for (int c = 1; c < NCLASS; ++c) m = fmaxf(m, logits[c]);
        float s = 0.f;
        for (int c = 0; c < NCLASS; ++c) s += __expf(logits[c] - m);
        red[0] = m;
        red[1] = __logf(s);
    }
    __syncthreads();
    if (tid < NCLASS)
        out[(size_t)node * NCLASS + tid] = logits[tid] - red[0] - red[1];
}

// ---------------- driver ----------------

extern "C" void kernel_launch(void* const* d_in, const int* in_sizes, int n_in,
                              void* d_out, int out_size, void* d_ws, size_t ws_size,
                              hipStream_t stream) {
    const float* x   = (const float*)d_in[0];
    const int*   ei  = (const int*)d_in[1];
    const float* W   = (const float*)d_in[2];
    const float* b   = (const float*)d_in[3];
    const float* fcW = (const float*)d_in[4];
    const float* fcb = (const float*)d_in[5];
    float* out = (float*)d_out;

    const int N = in_sizes[0] / NFEAT;     // 100000
    const int E = in_sizes[1] / 2;         // 1600000
    const int* src = ei;
    const int* dst = ei + E;

    const size_t n128 = (size_t)N * NFEAT;
    float* bufA = (float*)d_ws;            // h / aggregation output (ping-pong in place)
    float* bufT = bufA + n128;             // dense-transform output
    float* jk   = bufT + n128;             // running JK max
    float* dinv = jk + n128;
    int*   deg  = (int*)(dinv + N);

    const int n4 = (int)(n128 / 4);

    // symmetric GCN normalization
    k_init_deg <<<(N + 255) / 256, 256, 0, stream>>>(deg, N);
    k_deg_accum<<<(E + 255) / 256, 256, 0, stream>>>(dst, deg, E);
    k_dinv     <<<(N + 255) / 256, 256, 0, stream>>>(deg, dinv, N);
    k_zero4    <<<(n4 + 255) / 256, 256, 0, stream>>>((float4*)jk, n4);

    const float* hin = x;
    for (int l = 0; l < NLAYERS; ++l) {
        k_gemm_wmma<<<(N + 63) / 64, 256, 0, stream>>>(
            hin, W + (size_t)l * NFEAT * NFEAT, b + (size_t)l * NFEAT,
            dinv, bufT, bufA, N);
        k_scatter <<<(E + 7) / 8, 256, 0, stream>>>(bufT, src, dst, dinv, bufA, E);
        k_relu_jk <<<(n4 + 255) / 256, 256, 0, stream>>>((float4*)bufA, (float4*)jk, n4);
        hin = bufA;
    }
    k_fc_lsm<<<N, 64, 0, stream>>>(jk, fcW, fcb, out, N);
}
